// Block_6038724018435
// MI455X (gfx1250) — matrix operand order repaired and checked by reading, not compiled, not run on hardware
//
#include <hip/hip_runtime.h>
#include <hip/hip_bf16.h>

// ---------- types ----------
typedef __bf16 v16bf __attribute__((ext_vector_type(16)));
typedef __bf16 v8bf  __attribute__((ext_vector_type(8)));
typedef float  v8f   __attribute__((ext_vector_type(8)));
typedef unsigned short us8 __attribute__((ext_vector_type(8)));
typedef unsigned short us4 __attribute__((ext_vector_type(4)));
typedef unsigned int   u32x4 __attribute__((ext_vector_type(4)));
typedef unsigned int   u32x8 __attribute__((ext_vector_type(8)));

union Frag16 { v16bf v; v8bf h[2]; };

__device__ __forceinline__ unsigned short f32_to_bf16(float f) {
  union { float f; unsigned int u; } x; x.f = f;
  unsigned int u = x.u;
  u += 0x7FFFu + ((u >> 16) & 1u);   // round-to-nearest-even
  return (unsigned short)(u >> 16);
}

__device__ __forceinline__ v8f v8f_zero() {
  v8f z = {0.f,0.f,0.f,0.f,0.f,0.f,0.f,0.f};
  return z;
}

__device__ __forceinline__ float gelu_f(float x) {
  const float k = 0.7978845608028654f;
  return 0.5f * x * (1.f + tanhf(k * (x + 0.044715f * x * x * x)));
}

// ---------- TDM: DMA a 128-row x 32-col bf16 tile (row stride K elems) into
// LDS with 16B padding after each 64B row (-> LDS row stride 40 halves).
// D# per CDNA5 ISA. Issued once per wave;
// EXEC ignored; tracked by TENSORcnt. ----------
__device__ __forceinline__ void tdm_load_tile(unsigned lds_addr,
                                              const void* gptr,
                                              unsigned k_stride_elems) {
  unsigned long long ga = (unsigned long long)gptr;
  u32x4 g0;
  g0[0] = 1u;                                   // count=1, user descriptor
  g0[1] = lds_addr;                             // lds_addr [63:32]
  g0[2] = (unsigned)ga;                         // global_addr [95:64]
  g0[3] = ((unsigned)(ga >> 32) & 0x01FFFFFFu)  // global_addr [120:96]
        | 0x80000000u;                          // type=2 ("image") [127:126]
  u32x8 g1;
  // [17:16] data_size=1 (2B) | [20] pad_enable | [24:22] pad_interval=3 (16 dw)
  // | [31:25] pad_amount=3 (4 dwords = 16B)
  g1[0] = (1u << 16) | (1u << 20) | (3u << 22) | (3u << 25);
  g1[1] = 0u;                                   // tensor_dim0 lo16 (huge dim)
  g1[2] = 0x4000u;                              // tensor_dim0 = 1<<30 (no OOB)
  g1[3] = (32u << 16) | 0x10u;                  // tile_dim0=32 | tensor_dim1=1<<20
  g1[4] = 128u;                                 // tile_dim1=128, tile_dim2=0
  g1[5] = k_stride_elems;                       // tensor_dim0_stride lo32
  g1[6] = 0u;                                   // strides hi bits
  g1[7] = 0u;
  asm volatile("tensor_load_to_lds %0, %1" :: "s"(g0), "s"(g1) : "memory");
}

// ---------- weight convert + transpose: W[K,N] f32 -> Wt[N,K] bf16 ----------
__global__ void transpose_cvt_bf16(const float* __restrict__ W,
                                   unsigned short* __restrict__ Wt,
                                   int K, int N) {
  size_t i = (size_t)blockIdx.x * blockDim.x + threadIdx.x;
  if (i >= (size_t)K * N) return;
  int n = (int)(i % N);
  int k = (int)(i / N);
  Wt[(size_t)n * K + k] = f32_to_bf16(W[i]);
}

// ---------- LayerNorm over C=1024, bf16 output ----------
__global__ __launch_bounds__(256) void layernorm_bf16(
    const float* __restrict__ x, const float* __restrict__ w,
    const float* __restrict__ b, unsigned short* __restrict__ out) {
  __shared__ float red[16];
  const int row = blockIdx.x, tid = threadIdx.x;
  const float* xr = x + (size_t)row * 1024;
  float4 v = *(const float4*)(xr + tid * 4);
  float s  = v.x + v.y + v.z + v.w;
  float ss = v.x*v.x + v.y*v.y + v.z*v.z + v.w*v.w;
  #pragma unroll
  for (int m = 1; m < 32; m <<= 1) {
    s  += __shfl_xor(s,  m, 32);
    ss += __shfl_xor(ss, m, 32);
  }
  if ((tid & 31) == 0) { red[tid >> 5] = s; red[8 + (tid >> 5)] = ss; }
  __syncthreads();
  float ts = 0.f, tss = 0.f;
  #pragma unroll
  for (int i = 0; i < 8; ++i) { ts += red[i]; tss += red[8 + i]; }
  const float mean = ts * (1.f / 1024.f);
  const float var  = tss * (1.f / 1024.f) - mean * mean;
  const float rstd = rsqrtf(var + 1e-5f);
  const int c = tid * 4;
  float4 wv = *(const float4*)(w + c);
  float4 bv = *(const float4*)(b + c);
  us4 o;
  o[0] = f32_to_bf16((v.x - mean) * rstd * wv.x + bv.x);
  o[1] = f32_to_bf16((v.y - mean) * rstd * wv.y + bv.y);
  o[2] = f32_to_bf16((v.z - mean) * rstd * wv.z + bv.z);
  o[3] = f32_to_bf16((v.w - mean) * rstd * wv.w + bv.w);
  *(us4*)(out + (size_t)row * 1024 + c) = o;
}

// ---------- split qkv bf16 [B,T,3C] -> Q/K [B*H,T,64], Vt [B*H,64,T] ----------
__global__ void split_qkv(const unsigned short* __restrict__ qkv,
                          unsigned short* __restrict__ Q,
                          unsigned short* __restrict__ Kq,
                          unsigned short* __restrict__ Vt, size_t total) {
  size_t i = (size_t)blockIdx.x * blockDim.x + threadIdx.x;
  if (i >= total) return;
  int col = (int)(i % 3072);
  size_t rowi = i / 3072;
  int t = (int)(rowi & 1023);
  int b = (int)(rowi >> 10);
  int sec = col >> 10;
  int cc  = col & 1023;
  int h = cc >> 6, d = cc & 63;
  unsigned short v = qkv[i];
  size_t bh = (size_t)b * 16 + h;
  if (sec == 0)      Q [(bh * 1024 + t) * 64 + d] = v;
  else if (sec == 1) Kq[(bh * 1024 + t) * 64 + d] = v;
  else               Vt[(bh * 64 + d) * 1024 + t] = v;
}

// ---------- generic bf16 WMMA GEMM: C[M,N] = A[M,K] @ Bt[N,K]^T ----------
// TDM double-buffered tile staging; 128x128 block tile, K-step 32.
// EPI 1: +bias -> bf16 out     (QKV)
// EPI 2: +bias +resid -> f32   (O-proj, final proj)
// EPI 3: +bias, GELU -> bf16   (FC)
#define LDK 40

template <int EPI>
__global__ __launch_bounds__(256) void gemm_bf16_wmma(
    const unsigned short* __restrict__ A, const unsigned short* __restrict__ Bt,
    const float* __restrict__ bias, const float* __restrict__ resid,
    float* __restrict__ outF, unsigned short* __restrict__ outB,
    int M, int N, int K) {
  __shared__ unsigned short As[2][128 * LDK];
  __shared__ unsigned short Bs[2][128 * LDK];
  const int tid  = threadIdx.x;
  const int lane = tid & 31, wave = tid >> 5;
  const int wm = wave & 3, wn = wave >> 2;        // waves: 4 (M) x 2 (N)
  const int half = lane >> 4, ln = lane & 15;
  const int m0 = blockIdx.y * 128, n0 = blockIdx.x * 128;
  const int nt = K >> 5;

  v8f acc[2][4];
  #pragma unroll
  for (int i = 0; i < 2; ++i)
    #pragma unroll
    for (int j = 0; j < 4; ++j) acc[i][j] = v8f_zero();

  // prologue: wave 0 queues the first A/B tile pair into buffer 0
  if (wave == 0) {
    tdm_load_tile((unsigned)(size_t)(void*)&As[0][0], A  + (size_t)m0 * K, (unsigned)K);
    tdm_load_tile((unsigned)(size_t)(void*)&Bs[0][0], Bt + (size_t)n0 * K, (unsigned)K);
  }

  for (int i = 0; i < nt; ++i) {
    const int cur = i & 1;
    if (wave == 0) {
      if (i + 1 < nt) {
        const int kt2 = (i + 1) << 5;
        tdm_load_tile((unsigned)(size_t)(void*)&As[cur ^ 1][0],
                      A  + (size_t)m0 * K + kt2, (unsigned)K);
        tdm_load_tile((unsigned)(size_t)(void*)&Bs[cur ^ 1][0],
                      Bt + (size_t)n0 * K + kt2, (unsigned)K);
        // TDM is in-order per wave: <=2 outstanding means the pair feeding
        // buffer `cur` has landed; the next pair keeps streaming.
        __builtin_amdgcn_s_wait_tensorcnt(2);
      } else {
        __builtin_amdgcn_s_wait_tensorcnt(0);
      }
    }
    __syncthreads();

    Frag16 af[2], bf[4];
    #pragma unroll
    for (int mi = 0; mi < 2; ++mi) {
      const unsigned short* ap = &As[cur][(wm * 32 + mi * 16 + ln) * LDK];
      af[mi].h[0] = *(const v8bf*)(ap + half * 8);        // K = h*8..h*8+7
      af[mi].h[1] = *(const v8bf*)(ap + 16 + half * 8);   // K = 16+h*8..
    }
    #pragma unroll
    for (int nj = 0; nj < 4; ++nj) {
      const unsigned short* bp = &Bs[cur][(wn * 64 + nj * 16 + ln) * LDK];
      bf[nj].h[0] = *(const v8bf*)(bp + half * 8);        // K = h*8..h*8+7
      bf[nj].h[1] = *(const v8bf*)(bp + 16 + half * 8);
    }
    #pragma unroll
    for (int mi = 0; mi < 2; ++mi)
      #pragma unroll
      for (int nj = 0; nj < 4; ++nj)
        acc[mi][nj] = __builtin_amdgcn_wmma_f32_16x16x32_bf16(
            false, af[mi].v, false, bf[nj].v, (short)0, acc[mi][nj], false, false);
    __syncthreads();
  }

  // epilogue: D layout: VGPR j, lanes0-15 -> M=j, lanes16-31 -> M=8+j; N = lane&15
  #pragma unroll
  for (int mi = 0; mi < 2; ++mi) {
    #pragma unroll
    for (int nj = 0; nj < 4; ++nj) {
      const int c = n0 + wn * 64 + nj * 16 + ln;
      const float bv = bias[c];
      #pragma unroll
      for (int j = 0; j < 8; ++j) {
        const int r = m0 + wm * 32 + mi * 16 + j + half * 8;
        float v = acc[mi][nj][j] + bv;
        if (EPI == 2) v += resid[(size_t)r * N + c];
        if (EPI == 3) v = gelu_f(v);
        if (EPI == 2) outF[(size_t)r * N + c] = v;
        else          outB[(size_t)r * N + c] = f32_to_bf16(v);
      }
    }
  }
}

// ---------- flash attention (causal), WMMA bf16, fp32 softmax ----------
// Q,K: [B*H, T, 64] bf16 ; Vt: [B*H, 64, T] bf16 ; attn: [B*T, C] bf16
#define PLD 40
__global__ __launch_bounds__(128) void attn_wmma(
    const unsigned short* __restrict__ Q, const unsigned short* __restrict__ Kc,
    const unsigned short* __restrict__ Vt, unsigned short* __restrict__ attn) {
  __shared__ unsigned short Pl[4][16 * PLD];
  const int tid  = threadIdx.x;
  const int lane = tid & 31, wave = tid >> 5;
  const int half = lane >> 4, ln = lane & 15;
  const int bh = blockIdx.z * 16 + blockIdx.y;
  const int q0 = blockIdx.x * 64 + wave * 16;

  const unsigned short* Qb = Q  + (size_t)bh * 1024 * 64;
  const unsigned short* Kb = Kc + (size_t)bh * 1024 * 64;
  const unsigned short* Vb = Vt + (size_t)bh * 64 * 1024;

  // Q fragments held in registers (hs=64 -> 2 WMMA K-steps)
  Frag16 aq[2];
  const unsigned short* qr = Qb + (size_t)(q0 + ln) * 64;
  #pragma unroll
  for (int s = 0; s < 2; ++s) {
    aq[s].h[0] = *(const v8bf*)(qr + s * 32 + half * 8);
    aq[s].h[1] = *(const v8bf*)(qr + s * 32 + 16 + half * 8);
  }

  v8f oacc[4];
  float mrow[8], lrow[8];
  #pragma unroll
  for (int c2 = 0; c2 < 4; ++c2) oacc[c2] = v8f_zero();
  #pragma unroll
  for (int j = 0; j < 8; ++j) { mrow[j] = -1e30f; lrow[j] = 0.f; }

  const int kend = q0 + 16;
  for (int kc = 0; kc < kend; kc += 32) {
    // S = Q @ K^T for 16 q-rows x 32 kpos
    v8f sa[2];
    #pragma unroll
    for (int g = 0; g < 2; ++g) {
      sa[g] = v8f_zero();
      const unsigned short* kr = Kb + (size_t)(kc + g * 16 + ln) * 64;
      #pragma unroll
      for (int s = 0; s < 2; ++s) {
        Frag16 bk;
        bk.h[0] = *(const v8bf*)(kr + s * 32 + half * 8);
        bk.h[1] = *(const v8bf*)(kr + s * 32 + 16 + half * 8);
        sa[g] = __builtin_amdgcn_wmma_f32_16x16x32_bf16(
            false, aq[s].v, false, bk.v, (short)0, sa[g], false, false);
      }
    }

    // online softmax (rows live across 16-lane groups)
    float pv[2][8];
    #pragma unroll
    for (int j = 0; j < 8; ++j) {
      const int qrr = q0 + j + half * 8;
      float rmx = -1e30f;
      #pragma unroll
      for (int g = 0; g < 2; ++g) {
        float v = sa[g][j] * 0.125f;            // 1/sqrt(64)
        const int kp = kc + g * 16 + ln;
        if (kp > qrr) v = -1e30f;               // causal mask
        pv[g][j] = v;
        rmx = fmaxf(rmx, v);
      }
      #pragma unroll
      for (int m = 1; m < 16; m <<= 1) rmx = fmaxf(rmx, __shfl_xor(rmx, m, 32));
      const float mn = fmaxf(mrow[j], rmx);
      const float sc = __expf(mrow[j] - mn);
      mrow[j] = mn;
      float rs = 0.f;
      #pragma unroll
      for (int g = 0; g < 2; ++g) {
        float e = __expf(pv[g][j] - mn);
        pv[g][j] = e;
        rs += e;
      }
      #pragma unroll
      for (int m = 1; m < 16; m <<= 1) rs += __shfl_xor(rs, m, 32);
      lrow[j] = lrow[j] * sc + rs;
      #pragma unroll
      for (int c2 = 0; c2 < 4; ++c2) oacc[c2][j] *= sc;
    }

    // P (D-layout) -> LDS -> A-fragment layout
    unsigned short* pw = &Pl[wave][0];
    #pragma unroll
    for (int j = 0; j < 8; ++j) {
      pw[(j + half * 8) * PLD + ln]      = f32_to_bf16(pv[0][j]);
      pw[(j + half * 8) * PLD + 16 + ln] = f32_to_bf16(pv[1][j]);
    }
    asm volatile("s_wait_dscnt 0x0" ::: "memory");
    Frag16 ap;
    ap.h[0] = *(const v8bf*)(pw + ln * PLD + half * 8);
    ap.h[1] = *(const v8bf*)(pw + ln * PLD + 16 + half * 8);

    // O += P @ V  (V transposed: contiguous K per hs-row)
    #pragma unroll
    for (int c2 = 0; c2 < 4; ++c2) {
      Frag16 bv;
      const unsigned short* vr = Vb + (size_t)(c2 * 16 + ln) * 1024 + kc + half * 8;
      bv.h[0] = *(const v8bf*)(vr);
      bv.h[1] = *(const v8bf*)(vr + 16);
      oacc[c2] = __builtin_amdgcn_wmma_f32_16x16x32_bf16(
          false, ap.v, false, bv.v, (short)0, oacc[c2], false, false);
    }
  }

  // normalize + store bf16 into [B*T, C]
  #pragma unroll
  for (int c2 = 0; c2 < 4; ++c2) {
    #pragma unroll
    for (int j = 0; j < 8; ++j) {
      const float inv = 1.f / lrow[j];
      const size_t r = (size_t)blockIdx.z * 1024 + q0 + j + half * 8;
      attn[r * 1024 + blockIdx.y * 64 + c2 * 16 + ln] = f32_to_bf16(oacc[c2][j] * inv);
    }
  }
}

// ---------- launch ----------
extern "C" void kernel_launch(void* const* d_in, const int* in_sizes, int n_in,
                              void* d_out, int out_size, void* d_ws, size_t ws_size,
                              hipStream_t stream) {
  const float* x     = (const float*)d_in[0];
  const float* ln1w  = (const float*)d_in[1];
  const float* ln1b  = (const float*)d_in[2];
  const float* wqkv  = (const float*)d_in[3];
  const float* bqkv  = (const float*)d_in[4];
  const float* wo    = (const float*)d_in[5];
  const float* bo    = (const float*)d_in[6];
  const float* ln2w  = (const float*)d_in[7];
  const float* ln2b  = (const float*)d_in[8];
  const float* wfc   = (const float*)d_in[9];
  const float* bfc   = (const float*)d_in[10];
  const float* wproj = (const float*)d_in[11];
  const float* bproj = (const float*)d_in[12];
  float* out = (float*)d_out;

  const int Bb = 8, T = 1024, C = 1024, Hd = 4096, M = Bb * T;

  char* p = (char*)d_ws;
  auto alloc = [&](size_t bytes) -> void* {
    void* r = (void*)p;
    p += (bytes + 255) & ~(size_t)255;
    return r;
  };
  unsigned short* ln1h  = (unsigned short*)alloc((size_t)M * C * 2);
  unsigned short* wq_t  = (unsigned short*)alloc((size_t)3 * C * C * 2);
  unsigned short* wo_t  = (unsigned short*)alloc((size_t)C * C * 2);
  unsigned short* wfc_t = (unsigned short*)alloc((size_t)Hd * C * 2);
  unsigned short* wpr_t = (unsigned short*)alloc((size_t)C * Hd * 2);
  unsigned short* qkv   = (unsigned short*)alloc((size_t)M * 3 * C * 2);
  unsigned short* Qh    = (unsigned short*)alloc((size_t)M * C * 2);
  unsigned short* Kh    = (unsigned short*)alloc((size_t)M * C * 2);
  unsigned short* Vth   = (unsigned short*)alloc((size_t)M * C * 2);
  unsigned short* attn  = (unsigned short*)alloc((size_t)M * C * 2);
  float*          x2    = (float*)alloc((size_t)M * C * 4);
  unsigned short* h2    = (unsigned short*)alloc((size_t)M * C * 2);
  unsigned short* act   = (unsigned short*)alloc((size_t)M * Hd * 2);

  const int th = 256;
  transpose_cvt_bf16<<<(unsigned)(((size_t)C * 3 * C + th - 1) / th), th, 0, stream>>>(wqkv, wq_t, C, 3 * C);
  transpose_cvt_bf16<<<(unsigned)(((size_t)C * C + th - 1) / th),     th, 0, stream>>>(wo,   wo_t,  C, C);
  transpose_cvt_bf16<<<(unsigned)(((size_t)C * Hd + th - 1) / th),    th, 0, stream>>>(wfc,  wfc_t, C, Hd);
  transpose_cvt_bf16<<<(unsigned)(((size_t)Hd * C + th - 1) / th),    th, 0, stream>>>(wproj, wpr_t, Hd, C);

  layernorm_bf16<<<M, 256, 0, stream>>>(x, ln1w, ln1b, ln1h);

  gemm_bf16_wmma<1><<<dim3(3 * C / 128, M / 128), 256, 0, stream>>>(
      ln1h, wq_t, bqkv, nullptr, nullptr, qkv, M, 3 * C, C);

  {
    size_t tot = (size_t)M * 3 * C;
    split_qkv<<<(unsigned)((tot + th - 1) / th), th, 0, stream>>>(qkv, Qh, Kh, Vth, tot);
  }

  attn_wmma<<<dim3(T / 64, 16, Bb), 128, 0, stream>>>(Qh, Kh, Vth, attn);

  gemm_bf16_wmma<2><<<dim3(C / 128, M / 128), 256, 0, stream>>>(
      attn, wo_t, bo, x, x2, nullptr, M, C, C);

  layernorm_bf16<<<M, 256, 0, stream>>>(x2, ln2w, ln2b, h2);

  gemm_bf16_wmma<3><<<dim3(Hd / 128, M / 128), 256, 0, stream>>>(
      h2, wfc_t, bfc, nullptr, nullptr, act, M, Hd, C);

  gemm_bf16_wmma<2><<<dim3(C / 128, M / 128), 256, 0, stream>>>(
      act, wpr_t, bproj, x2, out, nullptr, M, C, Hd);
}
